// MultilayerGRU_2095944041069
// MI455X (gfx1250) — matrix operand order repaired and checked by reading, not compiled, not run on hardware
//
#include <hip/hip_runtime.h>
#include <hip/hip_bf16.h>

#define B_ 32
#define S_ 512
#define I_ 256
#define H_ 1024
#define O_ 256
#define L_ 2
#define NWG_REC 48

typedef __attribute__((ext_vector_type(16))) __bf16 v16bf;
typedef __attribute__((ext_vector_type(8)))  float  v8f;
typedef __attribute__((ext_vector_type(4)))  unsigned int u32x4;

union ABFrag { v16bf v; u32x4 q[2]; };

__device__ __forceinline__ unsigned short f2bf(float f) {
  unsigned u = __float_as_uint(f);
  u += 0x7fffu + ((u >> 16) & 1u);          // round-to-nearest-even
  return (unsigned short)(u >> 16);
}

// One 16x16 f32 tile accumulated over a K-segment with v_wmma_f32_16x16x32_bf16.
// A: row-major bf16, tile rows at Abase + mn*lda.  A-frag per ISA layout:
//   lanes 0-15: K = 0..7 & 16..23 ; lanes 16-31: K = 8..15 & 24..31 (per 32-K step)
// B: weight matrix row-major [N][K]; column n of B == row n of W (contiguous K).
//   same K layout as A: lanes 0-15: K = 0..7 & 16..23 ; lanes 16-31: K = 8..15 & 24..31.
__device__ __forceinline__ void mma_seg(v8f& acc,
    const unsigned short* __restrict__ Abase, long lda,
    const unsigned short* __restrict__ Bbase, int ldb, int Kseg, int lane) {
  const int mn  = lane & 15;
  const int ksA = (lane & 16) ? 8 : 0;
  const int ksB = (lane & 16) ? 8 : 0;
  const unsigned short* arow = Abase + (long)mn * lda;
  const unsigned short* brow = Bbase + (long)mn * ldb;
  #pragma unroll 4
  for (int k = 0; k < Kseg; k += 32) {
    ABFrag a, b;
    a.q[0] = *(const u32x4*)(arow + k + ksA);
    a.q[1] = *(const u32x4*)(arow + k + ksA + 16);
    b.q[0] = *(const u32x4*)(brow + k + ksB);
    b.q[1] = *(const u32x4*)(brow + k + ksB + 16);
    acc = __builtin_amdgcn_wmma_f32_16x16x32_bf16(false, a.v, false, b.v,
                                                  (short)0, acc, false, false);
  }
}

// Device-wide barrier: arrive counter + generation, L2 atomics, s_sleep spin.
__device__ __forceinline__ void grid_sync(unsigned* cnt, unsigned* gen,
                                          unsigned nwg, unsigned& my) {
  __syncthreads();
  my++;
  if (threadIdx.x == 0) {
    __threadfence();
    unsigned prev = atomicAdd(cnt, 1u);
    if (prev == nwg - 1u) {
      atomicSub(cnt, nwg);
      __threadfence();
      atomicAdd(gen, 1u);
    } else {
      while (atomicAdd(gen, 0u) < my) __builtin_amdgcn_s_sleep(1);
    }
  }
  __syncthreads();
  __threadfence();
}

__global__ void k_f2bf(const float* __restrict__ in,
                       unsigned short* __restrict__ out, long n) {
  long i = (long)blockIdx.x * blockDim.x + threadIdx.x;
  long stride = (long)gridDim.x * blockDim.x;
  for (; i < n; i += stride) out[i] = f2bf(in[i]);
}

// Persistent per-layer GRU kernel.  384 waves:
//   waves   0..127 : g gate  (phase1: x_t@Wg1^T held in regs; phase2: +(h*r)@Wg2^T, h update)
//   waves 128..255 : z gate  (x_t@Wz1^T + h@Wz2^T -> sigmoid -> zbuf)
//   waves 256..383 : r gate  (sigmoid -> hrbuf = bf16(h*r))
__global__ __launch_bounds__(256) void k_gru_layer(
    const unsigned short* __restrict__ xseq, long x_row_stride, long x_t_stride, int K1,
    const unsigned short* __restrict__ Wz1, const unsigned short* __restrict__ Wr1,
    const unsigned short* __restrict__ Wg1,
    const unsigned short* __restrict__ Wz2, const unsigned short* __restrict__ Wr2,
    const unsigned short* __restrict__ Wg2,
    const float* __restrict__ bz2, const float* __restrict__ br2,
    const float* __restrict__ bg2,
    const float* __restrict__ h_init, int layer,
    float* __restrict__ hbuf, unsigned short* __restrict__ hbf,
    float* __restrict__ zbuf, unsigned short* __restrict__ hrbuf,
    unsigned short* __restrict__ hs_out, float* __restrict__ dout_hidden,
    unsigned* cnt, unsigned* gen)
{
  const int lane = threadIdx.x & 31;
  const int w    = blockIdx.x * 8 + (threadIdx.x >> 5);
  unsigned my = 0;

  // init h (fp32 + bf16 copies)
  for (int i = blockIdx.x * blockDim.x + threadIdx.x; i < B_ * H_;
       i += gridDim.x * blockDim.x) {
    int b = i >> 10, h = i & (H_ - 1);
    float v = h_init[((long)b * L_ + layer) * H_ + h];
    hbuf[i] = v;
    hbf[i]  = f2bf(v);
  }
  grid_sync(cnt, gen, gridDim.x, my);

  const int role  = w >> 7;          // 0:g 1:z 2:r
  const int idx   = w & 127;
  const int mt    = idx & 1;         // M-tile (rows 0-15 / 16-31 of B=32)
  const int col0  = (idx >> 1) * 16; // N-tile origin
  const int mn    = lane & 15;
  const int mbase = mt * 16 + ((lane & 16) ? 8 : 0);
  const int col   = col0 + mn;

  const unsigned short* W1 = (role == 0) ? Wg1 : (role == 1) ? Wz1 : Wr1;
  const unsigned short* W2 = (role == 0) ? Wg2 : (role == 1) ? Wz2 : Wr2;
  const float* bias2       = (role == 0) ? bg2 : (role == 1) ? bz2 : br2;
  const float biasv = bias2[col];

  for (int t = 0; t < S_; ++t) {
    v8f acc = {};
    // ---- phase 1: x_t projection for all gates; +h@W2 for z,r ----
    const unsigned short* Ax =
        xseq + (long)t * x_t_stride + (long)(mt * 16) * x_row_stride;
    mma_seg(acc, Ax, x_row_stride, W1 + (long)col0 * K1, K1, K1, lane);
    if (role != 0) {
      mma_seg(acc, hbf + (long)(mt * 16) * H_, H_,
              W2 + (long)col0 * H_, H_, H_, lane);
      if (role == 1) {
        #pragma unroll
        for (int i = 0; i < 8; ++i) {
          float zv = 1.0f / (1.0f + __expf(-(acc[i] + biasv)));
          zbuf[(long)(mbase + i) * H_ + col] = zv;
        }
      } else {
        #pragma unroll
        for (int i = 0; i < 8; ++i) {
          float rv = 1.0f / (1.0f + __expf(-(acc[i] + biasv)));
          long o = (long)(mbase + i) * H_ + col;
          hrbuf[o] = f2bf(hbuf[o] * rv);
        }
      }
    }
    grid_sync(cnt, gen, gridDim.x, my);
    // ---- phase 2: g gate finishes with (h*r)@Wg2^T, then h update ----
    if (role == 0) {
      mma_seg(acc, hrbuf + (long)(mt * 16) * H_, H_,
              W2 + (long)col0 * H_, H_, H_, lane);
      #pragma unroll
      for (int i = 0; i < 8; ++i) {
        int row = mbase + i;
        long o = (long)row * H_ + col;
        float g  = tanhf(acc[i] + biasv);
        float z  = zbuf[o];
        float hn = z * hbuf[o] + (1.0f - z) * g;
        hbuf[o] = hn;
        unsigned short hb = f2bf(hn);
        hbf[o] = hb;
        hs_out[(long)t * (B_ * H_) + o] = hb;
        if (t == S_ - 2)
          dout_hidden[((long)row * L_ + layer) * H_ + col] = hn;
      }
    }
    grid_sync(cnt, gen, gridDim.x, my);
  }
}

// out[b,s,:] = hs1[s,b,:] @ Yw^T + Yb.  M=S*B=16384, N=O=256, K=H.
__global__ __launch_bounds__(256) void k_outproj(
    const unsigned short* __restrict__ Aseq, const unsigned short* __restrict__ Wy,
    const float* __restrict__ Yb, float* __restrict__ out)
{
  const int lane = threadIdx.x & 31;
  const int w  = blockIdx.x * 8 + (threadIdx.x >> 5);
  const int nw = gridDim.x * 8;
  const int mn = lane & 15;
  const int mb = (lane & 16) ? 8 : 0;
  const int MT = (S_ * B_) / 16;     // 1024
  const int NT = O_ / 16;            // 16
  for (int tile = w; tile < MT * NT; tile += nw) {
    int mt = tile / NT;
    int nt = tile - mt * NT;
    int col0 = nt * 16;
    v8f acc = {};
    mma_seg(acc, Aseq + (long)mt * 16 * H_, H_,
            Wy + (long)col0 * H_, H_, H_, lane);
    float bias = Yb[col0 + mn];
    #pragma unroll
    for (int i = 0; i < 8; ++i) {
      int mrow = mt * 16 + mb + i;
      int s = mrow >> 5;             // m = s*B + b, B=32
      int b = mrow & 31;
      out[((long)b * S_ + s) * O_ + col0 + mn] = acc[i] + bias;
    }
  }
}

extern "C" void kernel_launch(void* const* d_in, const int* in_sizes, int n_in,
                              void* d_out, int out_size, void* d_ws, size_t ws_size,
                              hipStream_t stream) {
  const float* x    = (const float*)d_in[0];
  const float* hst  = (const float*)d_in[1];
  const float* z1_0 = (const float*)d_in[2];
  const float* z2_0 = (const float*)d_in[3];
  const float* z2b0 = (const float*)d_in[4];
  const float* r1_0 = (const float*)d_in[5];
  const float* r2_0 = (const float*)d_in[6];
  const float* r2b0 = (const float*)d_in[7];
  const float* g1_0 = (const float*)d_in[8];
  const float* g2_0 = (const float*)d_in[9];
  const float* g2b0 = (const float*)d_in[10];
  const float* z1_1 = (const float*)d_in[11];
  const float* z2_1 = (const float*)d_in[12];
  const float* z2b1 = (const float*)d_in[13];
  const float* r1_1 = (const float*)d_in[14];
  const float* r2_1 = (const float*)d_in[15];
  const float* r2b1 = (const float*)d_in[16];
  const float* g1_1 = (const float*)d_in[17];
  const float* g2_1 = (const float*)d_in[18];
  const float* g2b1 = (const float*)d_in[19];
  const float* Yw   = (const float*)d_in[20];
  const float* Yb   = (const float*)d_in[21];

  char* ws = (char*)d_ws;
  size_t off = 0;
  auto take = [&](size_t bytes) -> char* {
    char* p = ws + off;
    off += (bytes + 255) & ~(size_t)255;
    return p;
  };
  unsigned short* xbf  = (unsigned short*)take((size_t)B_ * S_ * I_ * 2);
  unsigned short* wz10 = (unsigned short*)take((size_t)H_ * I_ * 2);
  unsigned short* wr10 = (unsigned short*)take((size_t)H_ * I_ * 2);
  unsigned short* wg10 = (unsigned short*)take((size_t)H_ * I_ * 2);
  unsigned short* wz20 = (unsigned short*)take((size_t)H_ * H_ * 2);
  unsigned short* wr20 = (unsigned short*)take((size_t)H_ * H_ * 2);
  unsigned short* wg20 = (unsigned short*)take((size_t)H_ * H_ * 2);
  unsigned short* wz11 = (unsigned short*)take((size_t)H_ * H_ * 2);
  unsigned short* wr11 = (unsigned short*)take((size_t)H_ * H_ * 2);
  unsigned short* wg11 = (unsigned short*)take((size_t)H_ * H_ * 2);
  unsigned short* wz21 = (unsigned short*)take((size_t)H_ * H_ * 2);
  unsigned short* wr21 = (unsigned short*)take((size_t)H_ * H_ * 2);
  unsigned short* wg21 = (unsigned short*)take((size_t)H_ * H_ * 2);
  unsigned short* wyb  = (unsigned short*)take((size_t)O_ * H_ * 2);
  unsigned short* hs0  = (unsigned short*)take((size_t)S_ * B_ * H_ * 2);
  unsigned short* hs1  = (unsigned short*)take((size_t)S_ * B_ * H_ * 2);
  float*          hbuf = (float*)take((size_t)B_ * H_ * 4);
  unsigned short* hbf  = (unsigned short*)take((size_t)B_ * H_ * 2);
  float*          zbuf = (float*)take((size_t)B_ * H_ * 4);
  unsigned short* hrb  = (unsigned short*)take((size_t)B_ * H_ * 2);
  unsigned*       bar  = (unsigned*)take(256);   // [cnt0,gen0,cnt1,gen1]

  auto cvt = [&](const float* src, unsigned short* dst, long n) {
    k_f2bf<<<512, 256, 0, stream>>>(src, dst, n);
  };
  cvt(x, xbf, (long)B_ * S_ * I_);
  cvt(z1_0, wz10, (long)H_ * I_);  cvt(r1_0, wr10, (long)H_ * I_);
  cvt(g1_0, wg10, (long)H_ * I_);
  cvt(z2_0, wz20, (long)H_ * H_);  cvt(r2_0, wr20, (long)H_ * H_);
  cvt(g2_0, wg20, (long)H_ * H_);
  cvt(z1_1, wz11, (long)H_ * H_);  cvt(r1_1, wr11, (long)H_ * H_);
  cvt(g1_1, wg11, (long)H_ * H_);
  cvt(z2_1, wz21, (long)H_ * H_);  cvt(r2_1, wr21, (long)H_ * H_);
  cvt(g2_1, wg21, (long)H_ * H_);
  cvt(Yw, wyb, (long)O_ * H_);

  hipMemsetAsync(bar, 0, 256, stream);

  float* dout_hid = (float*)d_out + (size_t)B_ * S_ * O_;

  // Layer 0: x_t rows at xbf + b*(S*I) + t*I, K1 = I
  k_gru_layer<<<NWG_REC, 256, 0, stream>>>(
      xbf, (long)S_ * I_, (long)I_, I_,
      wz10, wr10, wg10, wz20, wr20, wg20,
      z2b0, r2b0, g2b0,
      hst, 0,
      hbuf, hbf, zbuf, hrb,
      hs0, dout_hid, &bar[0], &bar[1]);

  // Layer 1: x_t rows at hs0 + t*(B*H) + b*H, K1 = H
  k_gru_layer<<<NWG_REC, 256, 0, stream>>>(
      hs0, (long)H_, (long)B_ * H_, H_,
      wz11, wr11, wg11, wz21, wr21, wg21,
      z2b1, r2b1, g2b1,
      hst, 1,
      hbuf, hbf, zbuf, hrb,
      hs1, dout_hid, &bar[2], &bar[3]);

  k_outproj<<<128, 256, 0, stream>>>(hs1, wyb, Yb, (float*)d_out);
}
